// XLSTMLayer_32366873543099
// MI455X (gfx1250) — matrix operand order repaired and checked by reading, not compiled, not run on hardware
//
#include <hip/hip_runtime.h>
#include <hip/hip_bf16.h>
#include <cstdint>

typedef __bf16 bf16_t;
typedef __attribute__((ext_vector_type(16))) __bf16 v16bf;
typedef __attribute__((ext_vector_type(8)))  float  v8f;

#define WMMA_BF16(a, b, c) \
  __builtin_amdgcn_wmma_f32_16x16x32_bf16(false, (a), false, (b), (short)0, (c), false, false)

// Problem dims
constexpr int kB   = 8;
constexpr int kS   = 2048;
constexpr int kDM  = 768;    // d_model
constexpr int kDH  = 1536;   // d_hidden
constexpr int kH   = 4;      // heads
constexpr int kDh  = 384;    // head dim
constexpr int kM   = kB * kS;      // 16384 rows
constexpr int kNcat = 6 * kDH;     // 9216 concatenated projection cols (i,f,o,k,v,q)
constexpr int kL   = 32;     // scan chunk length
constexpr int kRB  = 128;    // state rows per scan workgroup (3 blocks cover Dh=384)

// Dynamic LDS size for the scan kernel (306.25 KB < 320 KB/WGP)
constexpr size_t kScanLds =
    (size_t)kRB * kDh * 4 +      // C state (fp32)            196608
    (size_t)kL * kDh * 2 * 3 +   // Kb, Kt, Qb (bf16)          73728
    (size_t)kRB * kL * 2 +       // Wt (bf16)                   8192
    (size_t)kL * kL * 2 +        // Sm masked scores (bf16)     2048
    (size_t)kL * kRB * 4 * 2 +   // GL, OG (fp32)              32768
    (size_t)kL * 4 * 2;          // NQ, NORM                     256

__device__ inline float clamp20(float x) { return fminf(fmaxf(x, -20.f), 20.f); }
__device__ inline v8f v8f_zero() { v8f z = {0.f,0.f,0.f,0.f,0.f,0.f,0.f,0.f}; return z; }

// ---- CDNA5 async global->LDS copy (ASYNCcnt-tracked DMA, no VGPR round trip) ----
__device__ inline void async_load_b128(void* lds_dst, const void* gsrc) {
  // Generic shared pointer: low 32 bits are the wave-relative LDS offset.
  unsigned loff = (unsigned)(uintptr_t)lds_dst;
  asm volatile("global_load_async_to_lds_b128 %0, %1, off"
               :: "v"(loff), "v"(gsrc) : "memory");
}
__device__ inline void wait_async0() { asm volatile("s_wait_asynccnt 0x0" ::: "memory"); }
__device__ inline void wait_async6() { asm volatile("s_wait_asynccnt 0x6" ::: "memory"); }

// ---- WMMA operand loaders (generic pointers: global or LDS) ----
// A-matrix 16x32 bf16, source row-major [M][K] with stride (elements).
// Contiguous k order used here (same as load_tileBt): lane L: M = L%16; VGPR v holds K = (L/16)*16 + v*2 (+1)
__device__ inline v16bf load_tileA(const bf16_t* base, int stride, int lane) {
  union { v16bf v; unsigned u[8]; } r;
  const bf16_t* rowp = base + (lane & 15) * stride;
  const int khalf = (lane >> 4) & 1;
#pragma unroll
  for (int i = 0; i < 8; ++i) {
    const int k0 = (khalf << 4) + (i << 1);
    r.u[i] = *(const unsigned*)(rowp + k0);
  }
  return r.v;
}

// B-matrix 32x16 bf16 from "transposed" storage Bt[n][k] row-major (k contiguous).
// Contiguous k order used here (same as load_tileA): lane L: N = L%16; K offset = (L/16)*16, 2 per VGPR.
__device__ inline v16bf load_tileBt(const bf16_t* base, int stride, int lane) {
  union { v16bf v; unsigned u[8]; } r;
  const bf16_t* rowp = base + (lane & 15) * stride + (((lane >> 4) & 1) << 4);
#pragma unroll
  for (int i = 0; i < 8; ++i) r.u[i] = *(const unsigned*)(rowp + (i << 1));
  return r.v;
}

// Same as load_tileBt but converting fp32 source -> bf16 on the fly.
__device__ inline v16bf load_tileBt_f32(const float* base, int stride, int lane) {
  union { v16bf v; bf16_t e[16]; } r;
  const float* rowp = base + (lane & 15) * stride + (((lane >> 4) & 1) << 4);
#pragma unroll
  for (int i = 0; i < 16; ++i) r.e[i] = (bf16_t)rowp[i];
  return r.v;
}

// ============================ prep kernels ============================
__global__ void cvt_f32_bf16(const float* __restrict__ in, bf16_t* __restrict__ out, int n) {
  int i = blockIdx.x * 256 + threadIdx.x;
  if (i < n) out[i] = (bf16_t)in[i];
}

__global__ void build_bias(const float* __restrict__ bi, const float* __restrict__ bf_,
                           const float* __restrict__ bo, float* __restrict__ out) {
  int i = blockIdx.x * 256 + threadIdx.x;
  if (i < kNcat) {
    int seg = i / kDH, r = i - seg * kDH;
    float v = 0.f;
    if (seg == 0) v = bi[r];
    else if (seg == 1) v = bf_[r];
    else if (seg == 2) v = bo[r];
    out[i] = v;
  }
}

// ============================ GEMM kernel =============================
// C[M,N] = A[M,K] * Bw[N,K]^T (+bias).  A, Bw bf16 row-major; C fp32.
// 128x256 block, 8 waves (2x4), each wave 64x64 via 4x4 WMMA tiles
// (16 WMMAs per 8 tile gathers per k-step -> 2x LDS arithmetic intensity).
// Operands staged to LDS by async global->LDS b128 DMA, double buffered:
// prefetch k-step i+1 while WMMAs consume step i. 48 KB LDS per block.
__global__ __launch_bounds__(256) void gemm_bf16nt(
    const bf16_t* __restrict__ A, const bf16_t* __restrict__ Bw,
    float* __restrict__ Cout, const float* __restrict__ bias,
    int M, int N, int K) {
  __shared__ bf16_t Abuf[2][128 * 32];
  __shared__ bf16_t Bbuf[2][256 * 32];

  const int tid = threadIdx.x;
  const int lane = tid & 31, w = tid >> 5;
  const long blockM = blockIdx.x * 128, blockN = (long)blockIdx.y * 256;
  const int m_off = (w >> 2) * 64;   // 2 wave rows of 64
  const int n_off = (w & 3) * 64;    // 4 wave cols of 64

  v8f acc[4][4];
#pragma unroll
  for (int mi = 0; mi < 4; ++mi)
#pragma unroll
    for (int ni = 0; ni < 4; ++ni) acc[mi][ni] = v8f_zero();

  // A: 512 16B segments (2/thread); B: 1024 16B segments (4/thread) per k-step.
  auto stage = [&](int buf, int k0) {
#pragma unroll
    for (int s = 0; s < 2; ++s) {
      int seg = tid * 2 + s;
      int row = seg >> 2, part = (seg & 3) * 8;  // 8 bf16 = 16B
      async_load_b128(&Abuf[buf][row * 32 + part], A + (blockM + row) * K + k0 + part);
    }
#pragma unroll
    for (int s = 0; s < 4; ++s) {
      int seg = tid * 4 + s;
      int row = seg >> 2, part = (seg & 3) * 8;
      async_load_b128(&Bbuf[buf][row * 32 + part], Bw + (blockN + row) * K + k0 + part);
    }
  };

  const int nsteps = K / 32;
  stage(0, 0);
  for (int ks = 0; ks < nsteps; ++ks) {
    const int buf = ks & 1;
    if (ks + 1 < nsteps) {
      stage(buf ^ 1, (ks + 1) * 32);
      wait_async6();  // async loads retire in order: <=6 pending => prev stage landed
    } else {
      wait_async0();
    }
    __syncthreads();

    v16bf bt[4];
#pragma unroll
    for (int ni = 0; ni < 4; ++ni)
      bt[ni] = load_tileBt(&Bbuf[buf][(n_off + ni * 16) * 32], 32, lane);
#pragma unroll
    for (int mi = 0; mi < 4; ++mi) {
      v16bf a = load_tileA(&Abuf[buf][(m_off + mi * 16) * 32], 32, lane);
#pragma unroll
      for (int ni = 0; ni < 4; ++ni) acc[mi][ni] = WMMA_BF16(a, bt[ni], acc[mi][ni]);
    }
    __syncthreads();  // all reads done before this buffer is overwritten
  }

  const int nl = lane & 15;
  const int mo = ((lane >> 4) & 1) << 3;
#pragma unroll
  for (int mi = 0; mi < 4; ++mi)
#pragma unroll
    for (int ni = 0; ni < 4; ++ni) {
      int nn = (int)blockN + n_off + ni * 16 + nl;
      float bv = bias ? bias[nn] : 0.f;
#pragma unroll
      for (int r = 0; r < 8; ++r)
        Cout[(blockM + m_off + mi * 16 + mo + r) * N + nn] = acc[mi][ni][r] + bv;
    }
}

// ============================ scan kernel =============================
// One WG per (b, h, row-block of 128). Chunked mLSTM with per-row decay:
//   G_t[i] = prod_{s<=t} fg_s[i]   (within chunk)
//   C_t    = G_t * (C0 + sum_{s<=t} w_s k_s^T),  w_s[i] = i_s[i] v_s[i] / G_s[i]
//   read_t = G_t * (C0 q_t + sum_{s<=t} (q_t.k_s) w_s)
// State C (128x384 fp32, 196KB) lives in LDS (CDNA5: 320KB/WGP).
__global__ __launch_bounds__(256) void xlstm_scan(const float* __restrict__ P,
                                                  bf16_t* __restrict__ hs) {
  const int b = blockIdx.z, h = blockIdx.y, rb = blockIdx.x;
  const int i0 = rb * kRB;
  const int tid = threadIdx.x;
  const int lane = tid & 31, w = tid >> 5;

  extern __shared__ char smem_raw[];
  float*  Cst = (float*)smem_raw;                  // [kRB][kDh]
  bf16_t* Kb  = (bf16_t*)(Cst + kRB * kDh);        // [kL][kDh]  k, t-major
  bf16_t* Kt  = Kb + kL * kDh;                     // [kDh][kL]  k, j-major
  bf16_t* Qb  = Kt + kDh * kL;                     // [kL][kDh]  q, t-major
  bf16_t* Wt  = Qb + kL * kDh;                     // [kRB][kL]  w_s[i], i-major
  bf16_t* Sm  = Wt + kRB * kL;                     // [kL][kL]   masked scores
  float*  GL  = (float*)(Sm + kL * kL);            // [kL][kRB]  cumprod decay
  float*  OG  = GL + kL * kRB;                     // [kL][kRB]  sigmoid(o)
  float*  NQ  = OG + kL * kRB;                     // [kL]
  float*  NORM = NQ + kL;                          // [kL]

  const long prow = (long)b * kS;
  const int ci = 0 * kDH + h * kDh, cf = 1 * kDH + h * kDh, co = 2 * kDH + h * kDh,
            ck = 3 * kDH + h * kDh, cv = 4 * kDH + h * kDh, cq = 5 * kDH + h * kDh;

  for (int idx = tid; idx < kRB * kDh; idx += 256) Cst[idx] = 0.f;
  float n0 = 1.f, n1 = 1.f;  // normalizer carry for j = tid, tid+256
  __syncthreads();

  for (int c = 0; c < kS / kL; ++c) {
    const int t0 = c * kL;

    // Phase A: stage K, Q as bf16 (both orientations for K); prefetch next chunk
    for (int idx = tid; idx < kL * kDh; idx += 256) {
      int t = idx / kDh, j = idx - t * kDh;
      const float* rp = P + (prow + t0 + t) * (long)kNcat;
      float kk = rp[ck + j], qq = rp[cq + j];
      Kb[t * kDh + j] = (bf16_t)kk;
      Kt[j * kL + t]  = (bf16_t)kk;
      Qb[t * kDh + j] = (bf16_t)qq;
      if (c + 1 < kS / kL) {
        const float* rn = P + (prow + t0 + kL + t) * (long)kNcat;
        __builtin_prefetch(rn + ck + j, 0, 3);
        __builtin_prefetch(rn + cq + j, 0, 3);
        __builtin_prefetch(rn + cf + j, 0, 3);
        __builtin_prefetch(rn + ci + j, 0, 3);
      }
    }
    // Phase B: per-row gate prep (block rows only)
    if (tid < kRB) {
      const int igc = i0 + tid;
      float g = 1.f;
      for (int t = 0; t < kL; ++t) {
        const float* rp = P + (prow + t0 + t) * (long)kNcat;
        float fg = expf(clamp20(rp[cf + igc]));
        g *= fg;
        GL[t * kRB + tid] = g;
        float iv = expf(clamp20(rp[ci + igc]));
        Wt[tid * kL + t] = (bf16_t)(iv * rp[cv + igc] / g);
        OG[t * kRB + tid] = 1.f / (1.f + expf(-rp[co + igc]));
      }
    }
    if (tid < kL) NQ[tid] = 0.f;
    __syncthreads();

    // Phase C: n recurrence (full j range) + nq via wave butterfly + 1 atomic/wave
    {
      const int j1 = tid, j2 = tid + 256;
      for (int t = 0; t < kL; ++t) {
        const float* rp = P + (prow + t0 + t) * (long)kNcat;
        float fg = expf(clamp20(rp[cf + j1]));
        n0 = fg * n0 + expf(clamp20(rp[ci + j1])) * rp[ck + j1];
        float contrib = n0 * rp[cq + j1];
        if (j2 < kDh) {
          float fg2 = expf(clamp20(rp[cf + j2]));
          n1 = fg2 * n1 + expf(clamp20(rp[ci + j2])) * rp[ck + j2];
          contrib += n1 * rp[cq + j2];
        }
#pragma unroll
        for (int off = 16; off > 0; off >>= 1) contrib += __shfl_xor(contrib, off, 32);
        if (lane == 0) atomicAdd(&NQ[t], contrib);
      }
    }
    __syncthreads();
    if (tid < kL) NORM[tid] = fmaxf(fabsf(NQ[tid]), 1.f);

    // Phase E: S = Q K^T (32x32, K=384), causal mask, store bf16 (waves 0..3)
    if (w < 4) {
      const int tt = (w >> 1) * 16, ts = (w & 1) * 16;
      v8f acc = v8f_zero();
#pragma unroll
      for (int kk = 0; kk < kDh / 32; ++kk) {
        v16bf a  = load_tileA (Qb + tt * kDh + kk * 32, kDh, lane);
        v16bf bm = load_tileBt(Kb + ts * kDh + kk * 32, kDh, lane);
        acc = WMMA_BF16(a, bm, acc);
      }
      const int sl = ts + (lane & 15);
      const int tb = tt + (((lane >> 4) & 1) << 3);
#pragma unroll
      for (int r = 0; r < 8; ++r) {
        int tl = tb + r;
        Sm[tl * kL + sl] = (bf16_t)((sl <= tl) ? acc[r] : 0.f);
      }
    }
    __syncthreads();

    // Phase F: read = intra (Sm * Wt) + inter (Q * C0^T); wave w -> i-tile w
    {
      const int nt = w * 16;
      v8f acc0 = v8f_zero(), acc1 = v8f_zero();
      {
        v16bf bW = load_tileBt(Wt + nt * kL, kL, lane);
        v16bf a0 = load_tileA(Sm + 0 * kL, kL, lane);
        v16bf a1 = load_tileA(Sm + 16 * kL, kL, lane);
        acc0 = WMMA_BF16(a0, bW, acc0);
        acc1 = WMMA_BF16(a1, bW, acc1);
      }
#pragma unroll
      for (int kk = 0; kk < kDh / 32; ++kk) {
        v16bf bc = load_tileBt_f32(Cst + nt * kDh + kk * 32, kDh, lane);
        v16bf a0 = load_tileA(Qb + 0 * kDh + kk * 32, kDh, lane);
        v16bf a1 = load_tileA(Qb + 16 * kDh + kk * 32, kDh, lane);
        acc0 = WMMA_BF16(a0, bc, acc0);
        acc1 = WMMA_BF16(a1, bc, acc1);
      }
      // Phase G: h = og * G_t[i] * read / norm_t -> hs (bf16)
      const int iloc = nt + (lane & 15);
      const int tb = ((lane >> 4) & 1) << 3;
#pragma unroll
      for (int r = 0; r < 8; ++r) {
        int tl0 = tb + r, tl1 = 16 + tb + r;
        float v0 = OG[tl0 * kRB + iloc] * GL[tl0 * kRB + iloc] * acc0[r] / NORM[tl0];
        float v1 = OG[tl1 * kRB + iloc] * GL[tl1 * kRB + iloc] * acc1[r] / NORM[tl1];
        hs[(prow + t0 + tl0) * (long)kDH + h * kDh + i0 + iloc] = (bf16_t)v0;
        hs[(prow + t0 + tl1) * (long)kDH + h * kDh + i0 + iloc] = (bf16_t)v1;
      }
    }
    __syncthreads();  // all reads of C0 done before update

    // Phase H: C = G_L .* (C + Wt^T * K); wave w -> M-tile w (16 rows)
    {
      const int mt = w * 16;
      v16bf aW = load_tileA(Wt + mt * kL, kL, lane);
      const int tb = ((lane >> 4) & 1) << 3;
      float g31[8];
#pragma unroll
      for (int r = 0; r < 8; ++r) g31[r] = GL[(kL - 1) * kRB + mt + tb + r];
      const int ncol = lane & 15;
      for (int ntile = 0; ntile < kDh / 16; ++ntile) {
        float* cbase = Cst + mt * kDh + ntile * 16;
        v8f cacc;
#pragma unroll
        for (int r = 0; r < 8; ++r) cacc[r] = cbase[(tb + r) * kDh + ncol];
        v16bf bK = load_tileBt(Kt + (ntile * 16) * kL, kL, lane);
        cacc = WMMA_BF16(aW, bK, cacc);
#pragma unroll
        for (int r = 0; r < 8; ++r) cbase[(tb + r) * kDh + ncol] = cacc[r] * g31[r];
      }
    }
    __syncthreads();  // state consistent before next chunk
  }
}

// ============================ launcher ================================
extern "C" void kernel_launch(void* const* d_in, const int* in_sizes, int n_in,
                              void* d_out, int out_size, void* d_ws, size_t ws_size,
                              hipStream_t stream) {
  const float* x     = (const float*)d_in[0];
  const float* W_i   = (const float*)d_in[1];
  const float* b_i   = (const float*)d_in[2];
  const float* W_f   = (const float*)d_in[3];
  const float* b_f   = (const float*)d_in[4];
  const float* W_o   = (const float*)d_in[5];
  const float* b_o   = (const float*)d_in[6];
  const float* W_k   = (const float*)d_in[7];
  const float* W_v   = (const float*)d_in[8];
  const float* W_q   = (const float*)d_in[9];
  const float* W_out = (const float*)d_in[10];

  char* wsp = (char*)d_ws;
  auto take = [&](size_t bytes) {
    char* p = wsp;
    wsp += (bytes + 255) & ~(size_t)255;
    return p;
  };
  bf16_t* xb    = (bf16_t*)take((size_t)kM * kDM * 2);
  bf16_t* wcat  = (bf16_t*)take((size_t)kNcat * kDM * 2);
  float*  bcat  = (float*) take((size_t)kNcat * 4);
  bf16_t* woutb = (bf16_t*)take((size_t)kDM * kDH * 2);
  bf16_t* hsb   = (bf16_t*)take((size_t)kM * kDH * 2);
  float*  Pbuf  = (float*) take((size_t)kM * kNcat * 4);

  int n = kM * kDM;
  cvt_f32_bf16<<<(n + 255) / 256, 256, 0, stream>>>(x, xb, n);

  n = kDH * kDM;
  const float* Ws[6] = {W_i, W_f, W_o, W_k, W_v, W_q};
  for (int s = 0; s < 6; ++s)
    cvt_f32_bf16<<<(n + 255) / 256, 256, 0, stream>>>(Ws[s], wcat + (size_t)s * n, n);

  build_bias<<<(kNcat + 255) / 256, 256, 0, stream>>>(b_i, b_f, b_o, bcat);

  n = kDM * kDH;
  cvt_f32_bf16<<<(n + 255) / 256, 256, 0, stream>>>(W_out, woutb, n);

  // fused 6-way projection: P = X * Wcat^T + bias
  dim3 g1(kM / 128, kNcat / 256);
  gemm_bf16nt<<<g1, 256, 0, stream>>>(xb, wcat, Pbuf, bcat, kM, kNcat, kDM);

  // chunked scan: grid = (row-blocks, heads, batch)
  dim3 g2(kDh / kRB, kH, kB);
  xlstm_scan<<<g2, 256, kScanLds, stream>>>(Pbuf, hsb);

  // output projection: out = hs * W_out^T
  dim3 g3(kM / 128, kDM / 256);
  gemm_bf16nt<<<g3, 256, 0, stream>>>(hsb, woutb, (float*)d_out, nullptr, kM, kDM, kDH);
}
